// FitTorch_1425929142778
// MI455X (gfx1250) — matrix operand order repaired and checked by reading, not compiled, not run on hardware
//
#include <hip/hip_runtime.h>
#include <math.h>

// ---- static problem config (mirrors reference) ----
#define NATOMS 2000
#define NNBR   30
#define NPAIRS (NATOMS*NNBR)     // 60000
#define NCONF  8
#define NRAD   12
#define M3     20
#define DESCW  32                // 12 + 20
#define H1DIM  64
#define H2DIM  32
#define NTILES (NPAIRS/16)       // 3750 (exact)
#define TPW    5                 // tiles per wave; NTILES = 750*TPW exactly
#define NTASKS (NTILES/TPW)      // 750
#define CUTF   5.0f
#define ETA3   45.125f           // 1/(2*(2/19)^2)
#define PI_F   3.14159265358979323846f

typedef __attribute__((ext_vector_type(16))) _Float16 v16h;
typedef __attribute__((ext_vector_type(8)))  float    v8f;

__device__ __forceinline__ float sigmoidf(float x){ return 1.0f/(1.0f+__expf(-x)); }

// ---------------------------------------------------------------- zero output
__global__ void k_zero(float* out, int n){
  int i = blockIdx.x*blockDim.x + threadIdx.x;
  if (i < n) out[i] = 0.0f;
}

// ------------------------------------------------------ per-pair geometry+RBF
__global__ void k_geom(const float* __restrict__ x, const float* __restrict__ tx,
                       const int* __restrict__ ui, const int* __restrict__ uj,
                       float* __restrict__ u, float* __restrict__ r,
                       float* __restrict__ fc, float* __restrict__ desc){
  int p = blockIdx.x*blockDim.x + threadIdx.x;
  if (p >= NPAIRS) return;
  int i = ui[p], j = uj[p];
  float dx = x[i*3+0] - (tx[p*3+0] + x[j*3+0]);
  float dy = x[i*3+1] - (tx[p*3+1] + x[j*3+1]);
  float dz = x[i*3+2] - (tx[p*3+2] + x[j*3+2]);
  float rr = sqrtf(dx*dx + dy*dy + dz*dz);
  float inv = 1.0f / fmaxf(rr, 1e-12f);
  float ux = dx*inv, uy = dy*inv, uz = dz*inv;
  float fcv = 0.5f*(__cosf(PI_F*rr/CUTF) + 1.0f);
  u[p*3+0]=ux; u[p*3+1]=uy; u[p*3+2]=uz;
  r[p]=rr; fc[p]=fcv;
  float coef = sqrtf(2.0f/CUTF);
  #pragma unroll
  for (int n = 1; n <= NRAD; ++n){
    float an = (float)n * PI_F / CUTF;
    desc[p*DESCW + (n-1)] = fcv * coef * __sinf(an*rr) * inv;
  }
}

// -------------------------------------------------------- 3-body forward (d3)
__global__ __launch_bounds__(128) void k_3body_fwd(const float* __restrict__ u,
                                                   const float* __restrict__ fc,
                                                   float* __restrict__ desc){
  __shared__ float su[NNBR][3];
  __shared__ float sfc[NNBR];
  __shared__ float scos[NNBR][NNBR];
  int a = blockIdx.x, t = threadIdx.x;
  if (t < NNBR){
    int p = a*NNBR + t;
    su[t][0]=u[p*3+0]; su[t][1]=u[p*3+1]; su[t][2]=u[p*3+2];
    sfc[t]=fc[p];
  }
  __syncthreads();
  for (int idx = t; idx < NNBR*NNBR; idx += blockDim.x){
    int p = idx / NNBR, q = idx % NNBR;
    scos[p][q] = su[p][0]*su[q][0] + su[p][1]*su[q][1] + su[p][2]*su[q][2];
  }
  __syncthreads();
  for (int idx = t; idx < NNBR*M3; idx += blockDim.x){
    int p = idx / M3, m = idx % M3;
    float mu = -1.0f + 2.0f*(float)m/19.0f;
    float s = 0.0f;
    for (int q = 0; q < NNBR; ++q){
      if (q == p) continue;
      float d = scos[p][q] - mu;
      s += sfc[q] * __expf(-ETA3*d*d);
    }
    desc[(a*NNBR+p)*DESCW + NRAD + m] = s;
  }
}

// --------------------- pack f32 weights -> f16 B-fragments in ISA lane layout
// out fragment index: (nt*NKC + kc), per-lane 16 contiguous halfs (32B aligned)
template<int KDIM, int NOUT, bool TRANSB>
__global__ __launch_bounds__(32) void k_packB(const float* __restrict__ W,
                                              _Float16* __restrict__ out){
  constexpr int NKC = KDIM/32;
  int lane = threadIdx.x & 31;
  int idx  = blockIdx.x;              // nt*NKC + kc
  int nt = idx / NKC, kc = idx % NKC;
  int n = (lane & 15) + (nt<<4);
  #pragma unroll
  for (int e = 0; e < 16; ++e){
    int k = (kc<<5) + (e&7) + ((e>>3)<<4) + ((lane<16)?0:8); // ISA B layout
    float wv = TRANSB ? W[n*KDIM + k] : W[k*NOUT + n];
    out[((size_t)idx*32 + lane)*16 + e] = (_Float16)wv;
  }
}

// ------------------------------------------------- WMMA GEMM  Z = act(X)W + b
// A: 16 pairs x KDIM f16 (ISA A layout, vectorized float4 loads + cvt_pk),
// B: pre-packed f16 fragments loaded as v16h (b128 pairs), C/D f32.
// Each wave keeps all B fragments in VGPRs and sweeps TPW pair-tiles.
template<int KDIM, int NOUT, bool ACT>
__global__ __launch_bounds__(256) void k_gemm(const float* __restrict__ X,
                                              const _Float16* __restrict__ Bpack,
                                              const float* __restrict__ bias,
                                              float* __restrict__ Z){
  constexpr int NKC = KDIM/32;
  constexpr int NNT = NOUT/16;
  int wave = threadIdx.x >> 5;
  int lane = threadIdx.x & 31;
  int task = blockIdx.x * (blockDim.x >> 5) + wave;
  if (task >= NTASKS) return;               // wave-uniform: EXEC stays full
  int m = lane & 15;
  int kbase = (lane < 16) ? 0 : 8;

  // load all B fragments once per wave (reused across TPW tiles)
  const v16h* Bp = (const v16h*)Bpack;
  v16h B[NNT][NKC];
  #pragma unroll
  for (int nt = 0; nt < NNT; ++nt)
    #pragma unroll
    for (int kc = 0; kc < NKC; ++kc)
      B[nt][kc] = Bp[(size_t)(nt*NKC + kc)*32 + lane];

  for (int tt = 0; tt < TPW; ++tt){
    int tile = task*TPW + tt;
    int p = tile*16 + m;
    const float* row = X + (size_t)p*KDIM;
    if (tt + 1 < TPW)
      __builtin_prefetch(row + 16*KDIM, 0, 3);   // next tile's A rows

    v16h a[NKC];
    #pragma unroll
    for (int kc = 0; kc < NKC; ++kc){
      const float4* r0 = (const float4*)(row + (kc<<5) + kbase);       // k=kbase..+7
      const float4* r1 = (const float4*)(row + (kc<<5) + kbase + 16);  // k=kbase+16..+23
      float4 x0 = r0[0], x1 = r0[1], x2 = r1[0], x3 = r1[1];
      float av[16] = { x0.x,x0.y,x0.z,x0.w, x1.x,x1.y,x1.z,x1.w,
                       x2.x,x2.y,x2.z,x2.w, x3.x,x3.y,x3.z,x3.w };
      #pragma unroll
      for (int e = 0; e < 16; ++e){
        float v = av[e];
        if (ACT) v = sigmoidf(v);
        a[kc][e] = (_Float16)v;               // ISA A layout
      }
    }
    #pragma unroll
    for (int nt = 0; nt < NNT; ++nt){
      int n = (lane & 15) + (nt<<4);
      float bv = bias ? bias[n] : 0.0f;
      v8f c;
      #pragma unroll
      for (int v = 0; v < 8; ++v) c[v] = bv;
      #pragma unroll
      for (int kc = 0; kc < NKC; ++kc)
        c = __builtin_amdgcn_wmma_f32_16x16x32_f16(false, a[kc], false, B[nt][kc],
                                                   (short)0, c, false, false);
      #pragma unroll
      for (int v = 0; v < 8; ++v){
        int mm = v + ((lane<16)?0:8);                        // ISA C/D layout
        Z[(size_t)(tile*16+mm)*NOUT + n] = c[v];
      }
    }
  }
}

// -------------------- output layer: eij, energy scatter, g_z2 = dE/dz2 seed
__global__ __launch_bounds__(256) void k_out(const float* __restrict__ z2,
                                             const float* __restrict__ fc,
                                             const int* __restrict__ indices,
                                             const float* __restrict__ W3,
                                             const float* __restrict__ b3,
                                             float* __restrict__ energy,
                                             float* __restrict__ o_arr,
                                             float* __restrict__ gz2){
  __shared__ float se[NCONF];
  if (threadIdx.x < NCONF) se[threadIdx.x] = 0.0f;
  __syncthreads();
  int p = blockIdx.x*blockDim.x + threadIdx.x;
  if (p < NPAIRS){
    float o = b3[0];
    float h2v[H2DIM];
    #pragma unroll
    for (int n = 0; n < H2DIM; ++n){
      float h = sigmoidf(z2[p*H2DIM + n]);
      h2v[n] = h;
      o += h * W3[n];
    }
    float fcp = fc[p];
    atomicAdd(&se[indices[p]], o*fcp);
    o_arr[p] = o;
    #pragma unroll
    for (int n = 0; n < H2DIM; ++n)
      gz2[p*H2DIM + n] = fcp * W3[n] * h2v[n]*(1.0f - h2v[n]);
  }
  __syncthreads();
  if (threadIdx.x < NCONF) atomicAdd(&energy[threadIdx.x], se[threadIdx.x]);
}

// ------------------------ apply sigmoid' of layer1 to g_h1 (in place -> g_z1)
__global__ void k_act_bwd(const float* __restrict__ z1, float* __restrict__ g1){
  long idx = (long)blockIdx.x*blockDim.x + threadIdx.x;
  if (idx >= (long)NPAIRS*H1DIM) return;
  float h = sigmoidf(z1[idx]);
  g1[idx] *= h*(1.0f - h);
}

// -------------------------------------------------------- 3-body backward
__global__ __launch_bounds__(128) void k_3body_bwd(const float* __restrict__ u,
                                                   const float* __restrict__ fc,
                                                   const float* __restrict__ r,
                                                   const float* __restrict__ gdesc,
                                                   float* __restrict__ gdiff3,
                                                   float* __restrict__ gfc3){
  __shared__ float su[NNBR][3];
  __shared__ float sfc[NNBR];
  __shared__ float sg[NNBR][M3];
  __shared__ float scos[NNBR][NNBR];
  __shared__ float sT[NNBR][NNBR];
  __shared__ float sS[NNBR][NNBR];
  int a = blockIdx.x, t = threadIdx.x;
  if (t < NNBR){
    int p = a*NNBR + t;
    su[t][0]=u[p*3+0]; su[t][1]=u[p*3+1]; su[t][2]=u[p*3+2];
    sfc[t]=fc[p];
  }
  for (int idx = t; idx < NNBR*M3; idx += blockDim.x){
    int p = idx / M3, m = idx % M3;
    sg[p][m] = gdesc[(a*NNBR+p)*DESCW + NRAD + m];
  }
  __syncthreads();
  for (int idx = t; idx < NNBR*NNBR; idx += blockDim.x){
    int p = idx / NNBR, q = idx % NNBR;
    scos[p][q] = su[p][0]*su[q][0] + su[p][1]*su[q][1] + su[p][2]*su[q][2];
  }
  __syncthreads();
  for (int idx = t; idx < NNBR*NNBR; idx += blockDim.x){
    int p = idx / NNBR, q = idx % NNBR;
    float Tv = 0.0f, Sv = 0.0f;
    if (p != q){
      float cth = scos[p][q];
      #pragma unroll
      for (int m = 0; m < M3; ++m){
        float mu = -1.0f + 2.0f*(float)m/19.0f;
        float d = cth - mu;
        float g = __expf(-ETA3*d*d);
        float gm = sg[p][m];
        Sv += gm * g;
        Tv += gm * (-2.0f*ETA3*d) * g;
      }
    }
    sT[p][q] = Tv; sS[p][q] = Sv;
  }
  __syncthreads();
  if (t < NNBR){
    int p = t;
    float gx=0.0f, gy=0.0f, gz=0.0f, sfcg=0.0f;
    for (int q = 0; q < NNBR; ++q){
      if (q == p) continue;
      float coef = sfc[q]*sT[p][q] + sfc[p]*sT[q][p];
      gx += coef*su[q][0]; gy += coef*su[q][1]; gz += coef*su[q][2];
      sfcg += sS[q][p];          // dE/d fc(r_p) from 3-body weights
    }
    float dotv = gx*su[p][0] + gy*su[p][1] + gz*su[p][2];
    int pp = a*NNBR + p;
    float invr = 1.0f / r[pp];
    gdiff3[pp*3+0] = (gx - dotv*su[p][0])*invr;   // (I - uu^T)/r projection
    gdiff3[pp*3+1] = (gy - dotv*su[p][1])*invr;
    gdiff3[pp*3+2] = (gz - dotv*su[p][2])*invr;
    gfc3[pp] = sfcg;
  }
}

// --------------------------- radial backward + force scatter (F = -dE/dx)
__global__ void k_final(const float* __restrict__ u, const float* __restrict__ r,
                        const float* __restrict__ fc, const float* __restrict__ gdesc,
                        const float* __restrict__ o_arr, const float* __restrict__ gfc3,
                        const float* __restrict__ gdiff3,
                        const int* __restrict__ ui, const int* __restrict__ uj,
                        float* __restrict__ forces){
  int p = blockIdx.x*blockDim.x + threadIdx.x;
  if (p >= NPAIRS) return;
  float rr = r[p], fcp = fc[p];
  float invr = 1.0f/rr;
  float gfc = o_arr[p] + gfc3[p];     // direct eij=o*fc term + 3-body weight term
  float gr  = 0.0f;
  float coef = sqrtf(2.0f/CUTF);
  #pragma unroll
  for (int n = 1; n <= NRAD; ++n){
    float an = (float)n * PI_F / CUTF;
    float s = __sinf(an*rr), c = __cosf(an*rr);
    float grb = gdesc[p*DESCW + (n-1)];
    gfc += grb * coef * s * invr;
    gr  += grb * fcp * coef * (an*c*invr - s*invr*invr);
  }
  gr += gfc * (-0.5f*(PI_F/CUTF)*__sinf(PI_F*rr/CUTF));   // dfc/dr chain
  float gdx = gdiff3[p*3+0] + gr*u[p*3+0];
  float gdy = gdiff3[p*3+1] + gr*u[p*3+1];
  float gdz = gdiff3[p*3+2] + gr*u[p*3+2];
  int i = ui[p], j = uj[p];
  atomicAdd(&forces[i*3+0], -gdx); atomicAdd(&forces[i*3+1], -gdy); atomicAdd(&forces[i*3+2], -gdz);
  atomicAdd(&forces[j*3+0],  gdx); atomicAdd(&forces[j*3+1],  gdy); atomicAdd(&forces[j*3+2],  gdz);
}

// ---------------------------------------------------------------- launcher
extern "C" void kernel_launch(void* const* d_in, const int* in_sizes, int n_in,
                              void* d_out, int out_size, void* d_ws, size_t ws_size,
                              hipStream_t stream){
  const float* x  = (const float*)d_in[0];
  const float* tx = (const float*)d_in[1];
  const int* indices = (const int*)d_in[3];
  const int* ui = (const int*)d_in[6];
  const int* uj = (const int*)d_in[7];
  const float* W1 = (const float*)d_in[8];
  const float* b1 = (const float*)d_in[9];
  const float* W2 = (const float*)d_in[10];
  const float* b2 = (const float*)d_in[11];
  const float* W3 = (const float*)d_in[12];
  const float* b3 = (const float*)d_in[13];

  float* out    = (float*)d_out;       // [0..7] energy, [8..6007] forces
  float* forces = out + NCONF;

  float* ws = (float*)d_ws;
  size_t off = 0;
  float* u      = ws + off; off += (size_t)NPAIRS*3;
  float* rarr   = ws + off; off += NPAIRS;
  float* fcarr  = ws + off; off += NPAIRS;
  float* desc   = ws + off; off += (size_t)NPAIRS*DESCW;
  float* z1     = ws + off; off += (size_t)NPAIRS*H1DIM;
  float* z2     = ws + off; off += (size_t)NPAIRS*H2DIM;
  float* o_arr  = ws + off; off += NPAIRS;
  float* gz2    = ws + off; off += (size_t)NPAIRS*H2DIM;
  float* g1     = ws + off; off += (size_t)NPAIRS*H1DIM;
  float* gdesc  = ws + off; off += (size_t)NPAIRS*DESCW;
  float* gdiff3 = ws + off; off += (size_t)NPAIRS*3;
  float* gfc3   = ws + off; off += NPAIRS;
  off = (off + 7) & ~(size_t)7;                      // 32B align for v16h loads
  _Float16* Bp1  = (_Float16*)(ws + off); off += (DESCW*H1DIM)/2;  // fwd L1
  _Float16* Bp2  = (_Float16*)(ws + off); off += (H1DIM*H2DIM)/2;  // fwd L2
  _Float16* Bp2T = (_Float16*)(ws + off); off += (H2DIM*H1DIM)/2;  // bwd L2
  _Float16* Bp1T = (_Float16*)(ws + off); off += (H1DIM*DESCW)/2;  // bwd L1

  const int TB = 256;
  int nout = NCONF + NATOMS*3;
  k_zero<<<(nout+TB-1)/TB, TB, 0, stream>>>(out, nout);

  // pack all 4 weight configurations into f16 B fragments (ISA lane layout)
  k_packB<DESCW, H1DIM, false><<<(H1DIM/16)*(DESCW/32), 32, 0, stream>>>(W1, Bp1);
  k_packB<H1DIM, H2DIM, false><<<(H2DIM/16)*(H1DIM/32), 32, 0, stream>>>(W2, Bp2);
  k_packB<H2DIM, H1DIM, true ><<<(H1DIM/16)*(H2DIM/32), 32, 0, stream>>>(W2, Bp2T);
  k_packB<H1DIM, DESCW, true ><<<(DESCW/16)*(H1DIM/32), 32, 0, stream>>>(W1, Bp1T);

  k_geom<<<(NPAIRS+TB-1)/TB, TB, 0, stream>>>(x, tx, ui, uj, u, rarr, fcarr, desc);
  k_3body_fwd<<<NATOMS, 128, 0, stream>>>(u, fcarr, desc);

  int gemm_grid = (NTASKS + 7)/8;   // 8 waves (256 threads) per block
  // z1 = desc @ W1 + b1
  k_gemm<DESCW, H1DIM, false><<<gemm_grid, 256, 0, stream>>>(desc, Bp1, b1, z1);
  // z2 = sigmoid(z1) @ W2 + b2
  k_gemm<H1DIM, H2DIM, true ><<<gemm_grid, 256, 0, stream>>>(z1, Bp2, b2, z2);
  // output layer + energy + g_z2
  k_out<<<(NPAIRS+TB-1)/TB, TB, 0, stream>>>(z2, fcarr, indices, W3, b3, out, o_arr, gz2);
  // g_h1 = g_z2 @ W2^T
  k_gemm<H2DIM, H1DIM, false><<<gemm_grid, 256, 0, stream>>>(gz2, Bp2T, nullptr, g1);
  // g_z1 = g_h1 * h1*(1-h1)
  long nact = (long)NPAIRS*H1DIM;
  k_act_bwd<<<(int)((nact+TB-1)/TB), TB, 0, stream>>>(z1, g1);
  // g_desc = g_z1 @ W1^T
  k_gemm<H1DIM, DESCW, false><<<gemm_grid, 256, 0, stream>>>(g1, Bp1T, nullptr, gdesc);
  // 3-body backward
  k_3body_bwd<<<NATOMS, 128, 0, stream>>>(u, fcarr, rarr, gdesc, gdiff3, gfc3);
  // radial backward + force scatter
  k_final<<<(NPAIRS+TB-1)/TB, TB, 0, stream>>>(u, rarr, fcarr, gdesc, o_arr, gfc3,
                                               gdiff3, ui, uj, forces);
}
